// ScaledDotProductAttention_77137612636559
// MI455X (gfx1250) — matrix operand order repaired and checked by reading, not compiled, not run on hardware
//
#include <hip/hip_runtime.h>

#define BB 8
#define LL 2048
#define DD 64
#define KT 64   // keys per tile
#define MT 64   // query rows per workgroup (4 waves x 16)
#define NTILE (LL / KT)
#define PAD 72  // padded LDS row (bf16 elems): 144B, 16B-aligned, bank-friendly

typedef __bf16 bf16_t;
typedef bf16_t v16bf __attribute__((ext_vector_type(16)));
typedef bf16_t v8bf  __attribute__((ext_vector_type(8)));
typedef bf16_t v4bf  __attribute__((ext_vector_type(4)));
typedef float  v8f   __attribute__((ext_vector_type(8)));
typedef float  v4f   __attribute__((ext_vector_type(4)));

// fold 1/sqrt(64) and log2(e) into Q so P = exp2(S) is a bare v_exp_f32
#define QSCALE (0.125f * 1.44269504088896340736f)

#if defined(__gfx1250__) && __has_builtin(__builtin_amdgcn_global_load_async_to_lds_b128)
#define USE_ASYNC 1
#else
#define USE_ASYNC 0
#endif

#if USE_ASYNC
// Builtin parameter types (from hipcc diagnostic): pointers to 16B int vectors,
// src in addrspace(1) (printed "__device__"), dst in addrspace(3) (LDS).
typedef int v4i_vs __attribute__((vector_size(16)));
typedef __attribute__((address_space(1))) v4i_vs gv4i;
typedef __attribute__((address_space(3))) v4i_vs lv4i;

__device__ __forceinline__ void async_cp16(const bf16_t* g, bf16_t* l) {
    // 16B per lane, memory -> LDS, tracked by ASYNCcnt
    __builtin_amdgcn_global_load_async_to_lds_b128((gv4i*)g, (lv4i*)l, 0, 0);
}
__device__ __forceinline__ void wait_async_all() {
#if __has_builtin(__builtin_amdgcn_s_wait_asynccnt)
    __builtin_amdgcn_s_wait_asynccnt(0);
#else
    asm volatile("s_wait_asynccnt 0x0" ::: "memory");
#endif
}
#else
__device__ __forceinline__ void wait_async_all() {}
#endif

// ---------------------------------------------------------------------------
// Pre-pass: K (f32, [b][n][d]) -> bf16 row-major; V -> bf16 transposed [b][d][n]
// ---------------------------------------------------------------------------
__global__ __launch_bounds__(256)
void convert_kv_kernel(const float* __restrict__ k, const float* __restrict__ v,
                       bf16_t* __restrict__ kw, bf16_t* __restrict__ vw)
{
    size_t e    = (size_t)blockIdx.x * blockDim.x + threadIdx.x; // float4 id
    size_t base = e * 4;                                         // element id
    const int    d  = (int)(base & (DD - 1));
    const size_t bn = base / DD;           // b*L + n
    const int    b  = (int)(bn >> 11);     // / L
    const int    n  = (int)(bn & (LL - 1));

    v4f kf = *(const v4f*)(k + base);
    v4f vf = *(const v4f*)(v + base);
    v4bf k4;
    k4[0] = (bf16_t)kf[0]; k4[1] = (bf16_t)kf[1];
    k4[2] = (bf16_t)kf[2]; k4[3] = (bf16_t)kf[3];
    *(v4bf*)(kw + base) = k4;
    #pragma unroll
    for (int j = 0; j < 4; ++j)
        vw[((size_t)b * DD + d + j) * LL + n] = (bf16_t)vf[j];
}

// ---------------------------------------------------------------------------
// Main attention kernel: bf16 K/V from workspace, async double-buffered tiles
// ---------------------------------------------------------------------------
__global__ __launch_bounds__(128, 1)
void attn_wmma_async_kernel(const float*  __restrict__ q,
                            const bf16_t* __restrict__ kw,
                            const bf16_t* __restrict__ vw,
                            const int*    __restrict__ qm,
                            const int*    __restrict__ km,
                            float*        __restrict__ out)
{
    __shared__ __attribute__((aligned(16))) bf16_t kS[2][KT][PAD]; // [buf][key][d]
    __shared__ __attribute__((aligned(16))) bf16_t vS[2][DD][PAD]; // [buf][d][key]
    __shared__ __attribute__((aligned(16))) bf16_t pS[4][16][PAD]; // [wave][m][key]

    const int tid  = threadIdx.x;
    const int wave = tid >> 5;
    const int lane = tid & 31;
    const int half = lane >> 4;
    const int l15  = lane & 15;

    const int b    = blockIdx.x >> 5;
    const int q0   = (blockIdx.x & 31) * MT;
    const int qrow = q0 + wave * 16 + l15;

    const bf16_t* kwb = kw + (size_t)b * LL * DD; // [n][d]
    const bf16_t* vwb = vw + (size_t)b * DD * LL; // [d][n]
    const int*    kmb = km + (size_t)b * LL;

    // ---- Q A-fragments (16x32 bf16), scale+log2e folded in ---------------
    v16bf aq[2];
    {
        const float* qp = q + ((size_t)b * LL + qrow) * DD;
        #pragma unroll
        for (int f = 0; f < 2; ++f) {
            const int d0 = f * 32 + half * 8;
            v4f x0 = *(const v4f*)(qp + d0);
            v4f x1 = *(const v4f*)(qp + d0 + 4);
            v4f x2 = *(const v4f*)(qp + d0 + 16);
            v4f x3 = *(const v4f*)(qp + d0 + 20);
            v16bf a;
            #pragma unroll
            for (int j = 0; j < 4; ++j) {
                a[j]      = (bf16_t)(x0[j] * QSCALE);
                a[4 + j]  = (bf16_t)(x1[j] * QSCALE);
                a[8 + j]  = (bf16_t)(x2[j] * QSCALE);
                a[12 + j] = (bf16_t)(x3[j] * QSCALE);
            }
            aq[f] = a;
        }
    }

    const v8f vzero = {0.f, 0.f, 0.f, 0.f, 0.f, 0.f, 0.f, 0.f};
    v8f acc[4] = {vzero, vzero, vzero, vzero};
    float rs[8] = {0.f, 0.f, 0.f, 0.f, 0.f, 0.f, 0.f, 0.f};

    // ---- tile copy: 8 x 16B per thread (4 K rows-of-8, 4 V rows-of-8) ----
    auto issue_tile = [&](int t, int buf) {
        const int kt0 = t * KT;
        #pragma unroll
        for (int i = 0; i < 4; ++i) {
            const int e  = tid + i * 128;   // 0..511
            const int n  = e >> 3;
            const int db = (e & 7) * 8;
#if USE_ASYNC
            async_cp16(kwb + (size_t)(kt0 + n) * DD + db, &kS[buf][n][db]);
#else
            *(v8bf*)&kS[buf][n][db] = *(const v8bf*)(kwb + (size_t)(kt0 + n) * DD + db);
#endif
        }
        #pragma unroll
        for (int i = 0; i < 4; ++i) {
            const int e  = tid + i * 128;
            const int d  = e >> 3;
            const int nb = (e & 7) * 8;
#if USE_ASYNC
            async_cp16(vwb + (size_t)d * LL + kt0 + nb, &vS[buf][d][nb]);
#else
            *(v8bf*)&vS[buf][d][nb] = *(const v8bf*)(vwb + (size_t)d * LL + kt0 + nb);
#endif
        }
    };

    issue_tile(0, 0);

    for (int t = 0; t < NTILE; ++t) {
        const int cur = t & 1;
        wait_async_all();        // my copies for tile t have landed in LDS
        __syncthreads();         // everyone's copies have landed
        if (t + 1 < NTILE) issue_tile(t + 1, cur ^ 1); // overlap with compute

        const int kt0 = t * KT;

        // ---- S = Q.K^T (WMMA), P = km ? exp2(S) : 0 ----------------------
        #pragma unroll
        for (int ns = 0; ns < 4; ++ns) {
            const int kn = ns * 16 + l15;
            v8f c = vzero;
            #pragma unroll
            for (int f = 0; f < 2; ++f) {
                const int d0 = f * 32 + half * 8;
                v8bf blo = *(const v8bf*)&kS[cur][kn][d0];
                v8bf bhi = *(const v8bf*)&kS[cur][kn][d0 + 16];
                v16bf bk = __builtin_shufflevector(blo, bhi,
                    0,1,2,3,4,5,6,7,8,9,10,11,12,13,14,15);
                c = __builtin_amdgcn_wmma_f32_16x16x32_bf16(
                        false, aq[f], false, bk, (short)0, c, false, false);
            }
            const int kmv = kmb[kt0 + kn];
            #pragma unroll
            for (int r = 0; r < 8; ++r) {
                float p = kmv ? exp2f(c[r]) : 0.0f;   // bare v_exp_f32
                rs[r] += p;
                pS[wave][r + half * 8][ns * 16 + l15] = (bf16_t)p;
            }
        }

        // ---- O += P.V (WMMA) --------------------------------------------
        #pragma unroll
        for (int f = 0; f < 2; ++f) {
            const int n0 = f * 32 + half * 8;
            v8bf plo = *(const v8bf*)&pS[wave][l15][n0];
            v8bf phi = *(const v8bf*)&pS[wave][l15][n0 + 16];
            v16bf pa = __builtin_shufflevector(plo, phi,
                0,1,2,3,4,5,6,7,8,9,10,11,12,13,14,15);
            #pragma unroll
            for (int dsb = 0; dsb < 4; ++dsb) {
                const int d   = dsb * 16 + l15;
                const int vn0 = f * 32 + half * 8;
                v8bf vlo = *(const v8bf*)&vS[cur][d][vn0];
                v8bf vhi = *(const v8bf*)&vS[cur][d][vn0 + 16];
                v16bf bv = __builtin_shufflevector(vlo, vhi,
                    0,1,2,3,4,5,6,7,8,9,10,11,12,13,14,15);
                acc[dsb] = __builtin_amdgcn_wmma_f32_16x16x32_bf16(
                        false, pa, false, bv, (short)0, acc[dsb], false, false);
            }
        }
    }

    // ---- epilogue: clamp(sum,1) normalize, q-mask, store -----------------
    const int* qmb = qm + (size_t)b * LL;
    float*     ob  = out + ((size_t)b * LL + q0 + wave * 16) * DD;
    #pragma unroll
    for (int r = 0; r < 8; ++r) {
        float s = rs[r];
        s += __shfl_xor(s, 1, 32);
        s += __shfl_xor(s, 2, 32);
        s += __shfl_xor(s, 4, 32);
        s += __shfl_xor(s, 8, 32);
        const int   m  = r + half * 8;
        const float qv = qmb[q0 + wave * 16 + m] ? 1.0f : 0.0f;
        const float sc = qv / fmaxf(s, 1.0f);
        #pragma unroll
        for (int dsb = 0; dsb < 4; ++dsb)
            ob[(size_t)m * DD + dsb * 16 + l15] = acc[dsb][r] * sc;
    }
}

// ---------------------------------------------------------------------------
// Fallback (no workspace): round-1 style, f32 staging + on-the-fly bf16 cvt
// ---------------------------------------------------------------------------
__global__ __launch_bounds__(128, 1)
void attn_wmma_fallback_kernel(const float* __restrict__ q,
                               const float* __restrict__ k,
                               const float* __restrict__ v,
                               const int*   __restrict__ qm,
                               const int*   __restrict__ km,
                               float*       __restrict__ out)
{
    __shared__ __attribute__((aligned(16))) bf16_t kS[KT][PAD];
    __shared__ __attribute__((aligned(16))) bf16_t vS[DD][PAD];
    __shared__ __attribute__((aligned(16))) bf16_t pS[4][16][PAD];

    const int tid  = threadIdx.x;
    const int wave = tid >> 5;
    const int lane = tid & 31;
    const int half = lane >> 4;
    const int l15  = lane & 15;

    const int b    = blockIdx.x >> 5;
    const int q0   = (blockIdx.x & 31) * MT;
    const int qrow = q0 + wave * 16 + l15;

    v16bf aq[2];
    {
        const float* qp = q + ((size_t)b * LL + qrow) * DD;
        #pragma unroll
        for (int f = 0; f < 2; ++f) {
            const int d0 = f * 32 + half * 8;
            v4f x0 = *(const v4f*)(qp + d0);
            v4f x1 = *(const v4f*)(qp + d0 + 4);
            v4f x2 = *(const v4f*)(qp + d0 + 16);
            v4f x3 = *(const v4f*)(qp + d0 + 20);
            v16bf a;
            #pragma unroll
            for (int j = 0; j < 4; ++j) {
                a[j]      = (bf16_t)(x0[j] * QSCALE);
                a[4 + j]  = (bf16_t)(x1[j] * QSCALE);
                a[8 + j]  = (bf16_t)(x2[j] * QSCALE);
                a[12 + j] = (bf16_t)(x3[j] * QSCALE);
            }
            aq[f] = a;
        }
    }

    const v8f vzero = {0.f, 0.f, 0.f, 0.f, 0.f, 0.f, 0.f, 0.f};
    v8f acc[4] = {vzero, vzero, vzero, vzero};
    float rs[8] = {0.f, 0.f, 0.f, 0.f, 0.f, 0.f, 0.f, 0.f};

    const float* kb  = k  + (size_t)b * LL * DD;
    const float* vb  = v  + (size_t)b * LL * DD;
    const int*   kmb = km + (size_t)b * LL;

    for (int kt0 = 0; kt0 < LL; kt0 += KT) {
        __syncthreads();
        if (kt0 + KT < LL) {
            __builtin_prefetch(kb + (size_t)(kt0 + KT) * DD + tid * 32, 0, 0);
            __builtin_prefetch(vb + (size_t)(kt0 + KT) * DD + tid * 32, 0, 0);
        }
        #pragma unroll
        for (int i = 0; i < 8; ++i) {
            const int e = tid + i * 128;
            const int n = e >> 4;
            const int d = (e & 15) << 2;
            v4f kf = *(const v4f*)(kb + (size_t)(kt0 + n) * DD + d);
            v4f vf = *(const v4f*)(vb + (size_t)(kt0 + n) * DD + d);
            v4bf k4;
            k4[0] = (bf16_t)kf[0]; k4[1] = (bf16_t)kf[1];
            k4[2] = (bf16_t)kf[2]; k4[3] = (bf16_t)kf[3];
            *(v4bf*)&kS[n][d] = k4;
            vS[d + 0][n] = (bf16_t)vf[0];
            vS[d + 1][n] = (bf16_t)vf[1];
            vS[d + 2][n] = (bf16_t)vf[2];
            vS[d + 3][n] = (bf16_t)vf[3];
        }
        __syncthreads();

        #pragma unroll
        for (int ns = 0; ns < 4; ++ns) {
            const int kn = ns * 16 + l15;
            v8f c = vzero;
            #pragma unroll
            for (int f = 0; f < 2; ++f) {
                const int d0 = f * 32 + half * 8;
                v8bf blo = *(const v8bf*)&kS[kn][d0];
                v8bf bhi = *(const v8bf*)&kS[kn][d0 + 16];
                v16bf bk = __builtin_shufflevector(blo, bhi,
                    0,1,2,3,4,5,6,7,8,9,10,11,12,13,14,15);
                c = __builtin_amdgcn_wmma_f32_16x16x32_bf16(
                        false, aq[f], false, bk, (short)0, c, false, false);
            }
            const int kmv = kmb[kt0 + kn];
            #pragma unroll
            for (int r = 0; r < 8; ++r) {
                float p = kmv ? exp2f(c[r]) : 0.0f;
                rs[r] += p;
                pS[wave][r + half * 8][ns * 16 + l15] = (bf16_t)p;
            }
        }
        #pragma unroll
        for (int f = 0; f < 2; ++f) {
            const int n0 = f * 32 + half * 8;
            v8bf plo = *(const v8bf*)&pS[wave][l15][n0];
            v8bf phi = *(const v8bf*)&pS[wave][l15][n0 + 16];
            v16bf pa = __builtin_shufflevector(plo, phi,
                0,1,2,3,4,5,6,7,8,9,10,11,12,13,14,15);
            #pragma unroll
            for (int dsb = 0; dsb < 4; ++dsb) {
                const int d   = dsb * 16 + l15;
                const int vn0 = f * 32 + half * 8;
                v8bf vlo = *(const v8bf*)&vS[d][vn0];
                v8bf vhi = *(const v8bf*)&vS[d][vn0 + 16];
                v16bf bv = __builtin_shufflevector(vlo, vhi,
                    0,1,2,3,4,5,6,7,8,9,10,11,12,13,14,15);
                acc[dsb] = __builtin_amdgcn_wmma_f32_16x16x32_bf16(
                        false, pa, false, bv, (short)0, acc[dsb], false, false);
            }
        }
    }

    const int* qmb = qm + (size_t)b * LL;
    float*     ob  = out + ((size_t)b * LL + q0 + wave * 16) * DD;
    #pragma unroll
    for (int r = 0; r < 8; ++r) {
        float s = rs[r];
        s += __shfl_xor(s, 1, 32);
        s += __shfl_xor(s, 2, 32);
        s += __shfl_xor(s, 4, 32);
        s += __shfl_xor(s, 8, 32);
        const int   m  = r + half * 8;
        const float qv = qmb[q0 + wave * 16 + m] ? 1.0f : 0.0f;
        const float sc = qv / fmaxf(s, 1.0f);
        #pragma unroll
        for (int dsb = 0; dsb < 4; ++dsb)
            ob[(size_t)m * DD + dsb * 16 + l15] = acc[dsb][r] * sc;
    }
}

extern "C" void kernel_launch(void* const* d_in, const int* in_sizes, int n_in,
                              void* d_out, int out_size, void* d_ws, size_t ws_size,
                              hipStream_t stream) {
    (void)in_sizes; (void)n_in; (void)out_size;
    const float* q  = (const float*)d_in[0];
    const float* k  = (const float*)d_in[1];
    const float* v  = (const float*)d_in[2];
    const int*   qm = (const int*)d_in[3];
    const int*   km = (const int*)d_in[4];
    float* out = (float*)d_out;

    const size_t elems = (size_t)BB * LL * DD;
    const size_t need  = 2 * elems * sizeof(bf16_t);   // bf16 K + bf16 V^T

    dim3 grid(BB * (LL / MT));
    dim3 block(128);

    if (ws_size >= need && d_ws != nullptr) {
        bf16_t* kw = (bf16_t*)d_ws;
        bf16_t* vw = kw + elems;
        hipLaunchKernelGGL(convert_kv_kernel, dim3((unsigned)(elems / 4 / 256)),
                           dim3(256), 0, stream, k, v, kw, vw);
        hipLaunchKernelGGL(attn_wmma_async_kernel, grid, block, 0, stream,
                           q, kw, vw, qm, km, out);
    } else {
        hipLaunchKernelGGL(attn_wmma_fallback_kernel, grid, block, 0, stream,
                           q, k, v, qm, km, out);
    }
}
